// HFet_59390807769272
// MI455X (gfx1250) — matrix operand order repaired and checked by reading, not compiled, not run on hardware
//
#include <hip/hip_runtime.h>
#include <hip/hip_bf16.h>
#include <math.h>

// ---------------- problem constants ----------------
constexpr int NSENT_ = 256;
constexpr int B_     = 512;
constexpr int S_     = 128;
constexpr int D_     = 1024;
constexpr int L_     = 10331;
constexpr int LAT_   = 101;
constexpr float NEG_ = -10000.0f;

// LDS tile strides (bf16 elements). 72*2B = 144B row stride:
// multiple of 16B (aligned b128 loads) and odd multiple of 16B granules
// (9 granules) -> 16 lanes hit distinct bank quads.
constexpr int AS_ = 72;

// ---------------- WMMA bf16 16x16x32 (split-fp32 / "bf16x3") ----------------
typedef float  v8f   __attribute__((ext_vector_type(8)));
typedef __bf16 v4bf  __attribute__((ext_vector_type(4)));
typedef __bf16 v8bf  __attribute__((ext_vector_type(8)));
typedef __bf16 v16bf __attribute__((ext_vector_type(16)));

__device__ __forceinline__ v8f wmma_bf16(v16bf a, v16bf b, v8f c) {
  // D(16x16,f32) = A(16x32,bf16) * B(32x16,bf16) + C
  return __builtin_amdgcn_wmma_f32_16x16x32_bf16(
      /*neg_a=*/false, a, /*neg_b=*/false, b,
      /*c_mod=*/(short)0, c, /*reuse_a=*/false, /*reuse_b=*/false);
}

__device__ __forceinline__ v8bf ld8(const __bf16* p) {
  return *(const v8bf*)p;  // 16B LDS load
}
__device__ __forceinline__ v16bf cat8(v8bf a, v8bf b) {
  return __builtin_shufflevector(a, b, 0, 1, 2, 3, 4, 5, 6, 7,
                                 8, 9, 10, 11, 12, 13, 14, 15);
}

// split-fp32: x ~= hi + lo with hi,lo bf16 (fp32-accumulated bf16x3 GEMM)
__device__ __forceinline__ void split_store4(float4 v, __bf16* hi, __bf16* lo) {
  v4bf h, l;
  h.x = (__bf16)v.x; l.x = (__bf16)(v.x - (float)h.x);
  h.y = (__bf16)v.y; l.y = (__bf16)(v.y - (float)h.y);
  h.z = (__bf16)v.z; l.z = (__bf16)(v.z - (float)h.z);
  h.w = (__bf16)v.w; l.w = (__bf16)(v.w - (float)h.w);
  *(v4bf*)hi = h;
  *(v4bf*)lo = l;
}

// A fragment 16x32 bf16 (ISA 16-bit A layout): lane(half,m=l16):
//   elems 0..7  -> K = kk + half*8 .. +7
//   elems 8..15 -> K = kk + 16 + half*8 .. +7
__device__ __forceinline__ v16bf frag_a(const __bf16* rowp, int kk, int half) {
  return cat8(ld8(rowp + kk + half * 8), ld8(rowp + kk + 16 + half * 8));
}
// B fragment 32x16 bf16: lane(half,n=l16): elems 0..7 -> K = kk + half*8 .. +7,
//   elems 8..15 -> K = kk + 16 + half*8 .. +7
__device__ __forceinline__ v16bf frag_b(const __bf16* rowp, int kk, int half) {
  return cat8(ld8(rowp + kk + half * 8), ld8(rowp + kk + 16 + half * 8));
}

// =====================================================================
// Kernel 1: fused attention-score GEMM (bf16x3).
//   score[b,s] = sum_e tanh( (x W1^T)[b,s,e] + extra[b,e] + dist[b,s]*wd[e] ) * Wo[e]
//               + (1 - mask[b,s]) * NEG
// One workgroup per b: M=128 rows (all s), E chunks of 64, K blocks of 64.
// 8 waves: wave w owns rows w*16..w*16+15, 4 n-tile accumulators each.
// =====================================================================
__global__ __launch_bounds__(256) void attn_score_kernel(
    const float* __restrict__ elmo, const int* __restrict__ gathers,
    const float* __restrict__ W1,   const float* __restrict__ Wo,
    const float* __restrict__ mask, const float* __restrict__ dist,
    const float* __restrict__ wd,   const float* __restrict__ extra,
    float* __restrict__ score_out)
{
  const int b    = blockIdx.x;
  const int t    = threadIdx.x;
  const int lane = t & 31;
  const int wave = t >> 5;      // 0..7
  const int half = lane >> 4;   // 0/1
  const int l16  = lane & 15;

  __shared__ __align__(16) __bf16 ldsAhi[128 * AS_];   // x tile hi
  __shared__ __align__(16) __bf16 ldsAlo[128 * AS_];   // x tile lo
  __shared__ __align__(16) __bf16 ldsBhi[64 * AS_];    // W1 tile [e][k] hi
  __shared__ __align__(16) __bf16 ldsBlo[64 * AS_];    // W1 tile [e][k] lo
  __shared__ float ldsDist[S_];

  const int g = gathers[b];
  const float* __restrict__ xrow = elmo + (size_t)g * S_ * D_;

  if (t < S_) ldsDist[t] = dist ? dist[(size_t)b * S_ + t] : 0.0f;

  float part[8];
#pragma unroll
  for (int i = 0; i < 8; ++i) part[i] = 0.0f;

  for (int e0 = 0; e0 < D_; e0 += 64) {
    v8f acc[4] = {};
    for (int k0 = 0; k0 < D_; k0 += 64) {
      __syncthreads();
      // ---- stage A: 128 rows x 64 k, split to bf16 hi/lo ----
#pragma unroll
      for (int i = 0; i < 8; ++i) {
        const int slot = t + i * 256;            // 2048 float4 slots
        const int row  = slot >> 4;              // 0..127
        const int c4   = (slot & 15) << 2;       // 0..60
        const float4 v = *(const float4*)(xrow + (size_t)row * D_ + k0 + c4);
        split_store4(v, &ldsAhi[row * AS_ + c4], &ldsAlo[row * AS_ + c4]);
      }
      // ---- stage B: 64 e x 64 k straight copy of W1[e][k], split ----
      {
        const int e  = t & 63;
        const int kb = (t >> 6) << 4;            // 0,16,32,48
        const float* src = W1 + (size_t)(e0 + e) * D_ + k0 + kb;
#pragma unroll
        for (int j = 0; j < 4; ++j) {
          const float4 v = *(const float4*)(src + j * 4);
          split_store4(v, &ldsBhi[e * AS_ + kb + j * 4],
                          &ldsBlo[e * AS_ + kb + j * 4]);
        }
      }
      __syncthreads();
      // ---- compute: 2 K-steps of 32, 4 n-tiles, 3 WMMA each (hh, lh, hl) ----
      const __bf16* Ah = &ldsAhi[(wave * 16 + l16) * AS_];
      const __bf16* Al = &ldsAlo[(wave * 16 + l16) * AS_];
#pragma unroll
      for (int kk = 0; kk < 64; kk += 32) {
        const v16bf ah = frag_a(Ah, kk, half);
        const v16bf al = frag_a(Al, kk, half);
#pragma unroll
        for (int nt = 0; nt < 4; ++nt) {
          const v16bf bh = frag_b(&ldsBhi[(nt * 16 + l16) * AS_], kk, half);
          const v16bf bl = frag_b(&ldsBlo[(nt * 16 + l16) * AS_], kk, half);
          acc[nt] = wmma_bf16(ah, bh, acc[nt]);
          acc[nt] = wmma_bf16(al, bh, acc[nt]);
          acc[nt] = wmma_bf16(ah, bl, acc[nt]);
        }
      }
    }
    // ---- epilogue for this 64-wide E chunk: tanh + dot with Wo ----
#pragma unroll
    for (int nt = 0; nt < 4; ++nt) {
      const int e = e0 + nt * 16 + l16;
      const float woE = Wo[e];
      const float exE = extra ? extra[(size_t)b * D_ + e] : 0.0f;
      const float wdE = wd ? wd[e] : 0.0f;
#pragma unroll
      for (int i = 0; i < 8; ++i) {
        const int sl = wave * 16 + i + 8 * half;          // local row (= s)
        const float v = acc[nt][i] + exE + ldsDist[sl] * wdE;
        part[i] += tanhf(v) * woE;
      }
    }
  }
  // ---- reduce over the 16 lanes of each half (N direction) ----
#pragma unroll
  for (int i = 0; i < 8; ++i) {
    float p = part[i];
    p += __shfl_xor(p, 1, 32);
    p += __shfl_xor(p, 2, 32);
    p += __shfl_xor(p, 4, 32);
    p += __shfl_xor(p, 8, 32);
    part[i] = p;
  }
  if (l16 == 0) {
#pragma unroll
    for (int i = 0; i < 8; ++i) {
      const int s = wave * 16 + i + 8 * half;
      const float m = mask[(size_t)b * S_ + s];
      score_out[(size_t)b * S_ + s] = part[i] + (1.0f - m) * NEG_;
    }
  }
}

// =====================================================================
// Kernel 2: masked softmax over S + attention-weighted sum -> repr[b, D]
// =====================================================================
__global__ __launch_bounds__(256) void softmax_repr_kernel(
    const float* __restrict__ elmo, const int* __restrict__ gathers,
    const float* __restrict__ score, float* __restrict__ repr)
{
  const int b = blockIdx.x;
  const int t = threadIdx.x;
  __shared__ float sc[S_];
  __shared__ float red[2];

  if (t < S_) sc[t] = score[(size_t)b * S_ + t];
  __syncthreads();
  if (t == 0) {
    float mx = -INFINITY;
    for (int s = 0; s < S_; ++s) mx = fmaxf(mx, sc[s]);
    red[0] = mx;
  }
  __syncthreads();
  if (t < S_) sc[t] = expf(sc[t] - red[0]);
  __syncthreads();
  if (t == 0) {
    float sm = 0.0f;
    for (int s = 0; s < S_; ++s) sm += sc[s];
    red[1] = sm;
  }
  __syncthreads();
  if (t < S_) sc[t] = sc[t] / red[1];
  __syncthreads();

  const float* __restrict__ xrow = elmo + (size_t)gathers[b] * S_ * D_;
  for (int d = t; d < D_; d += 256) {
    float a = 0.0f;
    for (int s = 0; s < S_; ++s) a += xrow[(size_t)s * D_ + d] * sc[s];
    repr[(size_t)b * D_ + d] = a;
  }
}

// =====================================================================
// Generic bf16x3 GEMM: C[m,n] = sum_k A[m,k] * W[n,k]  (+ scale*ep[m,n])
// SPLIT_A: A = concat(A0, A1) along k (each half stride K/2).
// Tile: 64 M x 128 N, KB=64. 8 waves = 4(m) x 2(n), 4 n-tiles each.
// =====================================================================
template <bool SPLIT_A, bool ADD_EP>
__global__ __launch_bounds__(256) void gemm_nt_kernel(
    const float* __restrict__ A0, const float* __restrict__ A1,
    const float* __restrict__ W,
    const float* __restrict__ ep, const float* __restrict__ epScale,
    float* __restrict__ Cout, int M, int N, int K)
{
  const int t    = threadIdx.x;
  const int lane = t & 31;
  const int wave = t >> 5;
  const int half = lane >> 4;
  const int l16  = lane & 15;
  const int wm   = wave & 3;   // m tile within block
  const int wn   = wave >> 2;  // n group (0/1), each covers 64 n

  __shared__ __align__(16) __bf16 ldsAhi[64 * AS_];
  __shared__ __align__(16) __bf16 ldsAlo[64 * AS_];
  __shared__ __align__(16) __bf16 ldsBhi[128 * AS_];
  __shared__ __align__(16) __bf16 ldsBlo[128 * AS_];

  const int m_base = blockIdx.y * 64;
  const int n_base = blockIdx.x * 128;
  const int lda = SPLIT_A ? (K >> 1) : K;

  v8f acc[4] = {};

  for (int k0 = 0; k0 < K; k0 += 64) {
    __syncthreads();
    // ---- stage A 64x64, split to bf16 hi/lo ----
    const float* Asrc = A0;
    int krel = k0;
    if (SPLIT_A && k0 >= (K >> 1)) { Asrc = A1; krel = k0 - (K >> 1); }
#pragma unroll
    for (int i = 0; i < 4; ++i) {
      const int slot = t + i * 256;           // 1024 float4 slots
      const int row  = slot >> 4;             // 0..63
      const int c4   = (slot & 15) << 2;
      const float4 v =
          *(const float4*)(Asrc + (size_t)(m_base + row) * lda + krel + c4);
      split_store4(v, &ldsAhi[row * AS_ + c4], &ldsAlo[row * AS_ + c4]);
    }
    // ---- stage B 128n x 64k straight copy of W[n][k], split ----
    {
      const int n  = t & 127;
      const int kb = (t >> 7) << 5;           // 0 or 32
      const int ng = n_base + n;
#pragma unroll
      for (int j = 0; j < 8; ++j) {
        float4 v = make_float4(0.f, 0.f, 0.f, 0.f);
        if (ng < N) v = *(const float4*)(W + (size_t)ng * K + k0 + kb + j * 4);
        split_store4(v, &ldsBhi[n * AS_ + kb + j * 4],
                        &ldsBlo[n * AS_ + kb + j * 4]);
      }
    }
    __syncthreads();
    // ---- compute ----
    const __bf16* Ah = &ldsAhi[(wm * 16 + l16) * AS_];
    const __bf16* Al = &ldsAlo[(wm * 16 + l16) * AS_];
#pragma unroll
    for (int kk = 0; kk < 64; kk += 32) {
      const v16bf ah = frag_a(Ah, kk, half);
      const v16bf al = frag_a(Al, kk, half);
#pragma unroll
      for (int nt = 0; nt < 4; ++nt) {
        const int brow = (wn * 64 + nt * 16 + l16) * AS_;
        const v16bf bh = frag_b(&ldsBhi[brow], kk, half);
        const v16bf bl = frag_b(&ldsBlo[brow], kk, half);
        acc[nt] = wmma_bf16(ah, bh, acc[nt]);
        acc[nt] = wmma_bf16(al, bh, acc[nt]);
        acc[nt] = wmma_bf16(ah, bl, acc[nt]);
      }
    }
  }
  // ---- epilogue ----
  const float scale = ADD_EP ? epScale[0] : 0.0f;
#pragma unroll
  for (int nt = 0; nt < 4; ++nt) {
    const int n = n_base + wn * 64 + nt * 16 + l16;
    if (n < N) {
#pragma unroll
      for (int i = 0; i < 8; ++i) {
        const int m = m_base + wm * 16 + i + 8 * half;
        float v = acc[nt][i];
        if (ADD_EP) v += scale * ep[(size_t)m * N + n];
        Cout[(size_t)m * N + n] = v;
      }
    }
  }
}

// =====================================================================
// latent = [men|ctx] @ W_f2l^T   ([B,101], K=2048)
// =====================================================================
__global__ __launch_bounds__(256) void f2l_kernel(
    const float* __restrict__ men, const float* __restrict__ ctx,
    const float* __restrict__ Wf2l, float* __restrict__ latent)
{
  const int b = blockIdx.x;
  const int t = threadIdx.x;
  __shared__ float fr[2 * D_];
  for (int k = t; k < D_; k += 256) {
    fr[k]      = men[(size_t)b * D_ + k];
    fr[D_ + k] = ctx[(size_t)b * D_ + k];
  }
  __syncthreads();
  if (t < LAT_) {
    const float* wrow = Wf2l + (size_t)t * (2 * D_);
    float a = 0.0f;
    for (int k = 0; k < 2 * D_; ++k) a += fr[k] * wrow[k];
    latent[(size_t)b * LAT_ + t] = a;
  }
}

// =====================================================================
// outputs_latent = latent @ W_l2l^T   ([B, L], K=101)
// Block: 128 l-rows staged in LDS, 64 b's per blockIdx.y, 2 b per pass.
// =====================================================================
__global__ __launch_bounds__(256) void l2l_kernel(
    const float* __restrict__ latent, const float* __restrict__ Wl2l,
    float* __restrict__ out_latent)
{
  const int t  = threadIdx.x;
  const int l0 = blockIdx.x * 128;
  const int b0 = blockIdx.y * 64;
  __shared__ float wt[128 * LAT_];
  __shared__ float lat2[2 * LAT_];

  for (int idx = t; idx < 128 * LAT_; idx += 256) {
    const int lr = idx / LAT_;
    const int k  = idx % LAT_;
    const int lg = l0 + lr;
    wt[idx] = (lg < L_) ? Wl2l[(size_t)lg * LAT_ + k] : 0.0f;
  }
  const int lr = t & 127;
  const int bh = t >> 7;  // 0/1
  for (int b = b0; b < b0 + 64; b += 2) {
    __syncthreads();
    for (int k = t; k < 2 * LAT_; k += 256) {
      const int bb = b + k / LAT_;
      lat2[k] = latent[(size_t)bb * LAT_ + (k % LAT_)];
    }
    __syncthreads();
    const int bb = b + bh;
    const float* lrow = &lat2[bh * LAT_];
    const float* wrow = &wt[lr * LAT_];
    float a = 0.0f;
    for (int k = 0; k < LAT_; ++k) a += lrow[k] * wrow[k];
    const int lg = l0 + lr;
    if (lg < L_) out_latent[(size_t)bb * L_ + lg] = a;
  }
}

// =====================================================================
// launch
// =====================================================================
extern "C" void kernel_launch(void* const* d_in, const int* in_sizes, int n_in,
                              void* d_out, int out_size, void* d_ws, size_t ws_size,
                              hipStream_t stream) {
  (void)in_sizes; (void)n_in; (void)out_size; (void)ws_size;
  const float* elmo      = (const float*)d_in[0];
  const float* men_mask  = (const float*)d_in[1];
  const float* ctx_mask  = (const float*)d_in[2];
  const float* dist      = (const float*)d_in[3];
  const int*   gathers   = (const int*)  d_in[4];
  const float* W_men_m   = (const float*)d_in[5];
  const float* W_men_o   = (const float*)d_in[6];
  const float* W_ctx_c   = (const float*)d_in[7];
  const float* W_ctx_m   = (const float*)d_in[8];
  const float* w_ctx_d   = (const float*)d_in[9];
  const float* W_ctx_o   = (const float*)d_in[10];
  const float* W_out     = (const float*)d_in[11];
  const float* W_f2l     = (const float*)d_in[12];
  const float* W_l2l     = (const float*)d_in[13];
  const float* lat_scale = (const float*)d_in[14];

  float* outputs    = (float*)d_out;                        // [B, L]
  float* out_latent = (float*)d_out + (size_t)B_ * L_;      // [B, L]

  // workspace layout (floats)
  float* ws       = (float*)d_ws;
  float* score    = ws;                                 // B*S
  float* men_repr = score + (size_t)B_ * S_;            // B*D
  float* ctx_repr = men_repr + (size_t)B_ * D_;         // B*D
  float* m_proj   = ctx_repr + (size_t)B_ * D_;         // B*D
  float* latent   = m_proj + (size_t)B_ * D_;           // B*LAT

  // 1) mention attention scores
  attn_score_kernel<<<B_, 256, 0, stream>>>(
      elmo, gathers, W_men_m, W_men_o, men_mask,
      nullptr, nullptr, nullptr, score);
  // 2) mention softmax + weighted sum
  softmax_repr_kernel<<<B_, 256, 0, stream>>>(elmo, gathers, score, men_repr);
  // 3) m_proj = men_repr @ W_ctx_m^T
  gemm_nt_kernel<false, false><<<dim3(D_ / 128, B_ / 64), 256, 0, stream>>>(
      men_repr, nullptr, W_ctx_m, nullptr, nullptr, m_proj, B_, D_, D_);
  // 4) context attention scores (extra + dist*w_d fused)
  attn_score_kernel<<<B_, 256, 0, stream>>>(
      elmo, gathers, W_ctx_c, W_ctx_o, ctx_mask,
      dist, w_ctx_d, m_proj, score);
  // 5) context softmax + weighted sum
  softmax_repr_kernel<<<B_, 256, 0, stream>>>(elmo, gathers, score, ctx_repr);
  // 6) latent = final @ W_f2l^T
  f2l_kernel<<<B_, 256, 0, stream>>>(men_repr, ctx_repr, W_f2l, latent);
  // 7) outputs_latent = latent @ W_l2l^T  (second tuple output)
  l2l_kernel<<<dim3((L_ + 127) / 128, B_ / 64), 256, 0, stream>>>(
      latent, W_l2l, out_latent);
  // 8) outputs = final @ W_out^T + latent_scalar * outputs_latent
  gemm_nt_kernel<true, true><<<dim3((L_ + 127) / 128, B_ / 64), 256, 0, stream>>>(
      men_repr, ctx_repr, W_out, out_latent, lat_scale, outputs, B_, L_, 2 * D_);
}
